// SpectralLSTMCell_29214367547628
// MI455X (gfx1250) — matrix operand order repaired and checked by reading, not compiled, not run on hardware
//
#include <hip/hip_runtime.h>

#define Hdim 256
#define Idim 128

typedef __attribute__((ext_vector_type(16))) __bf16 v16bf;
typedef __attribute__((ext_vector_type(8)))  __bf16 v8bf;
typedef __attribute__((ext_vector_type(8)))  float  v8f;

static __device__ __forceinline__ float sigf(float x) { return 1.0f / (1.0f + __expf(-x)); }

static __device__ __forceinline__ v8f wmma_bf16(v16bf a, v16bf b, v8f c) {
    return __builtin_amdgcn_wmma_f32_16x16x32_bf16(false, a, false, b, (short)0, c, false, false);
}

// B fragment (16x16x32 bf16): lane L holds column N = L%16, K = (L/16)*16 + 2v+{0,1}.
// Source stored row-major as [N][K] -> one contiguous 32B read per lane.
static __device__ __forceinline__ v16bf load_b_frag(const __bf16* __restrict__ M,
                                                    int row, int k0, int ldk, int hl) {
    return *reinterpret_cast<const v16bf*>(M + (size_t)row * ldk + k0 + hl * 16);
}

// A fragment (16x32 bf16) from an LDS strip: lane L holds row M = L%16;
// VGPR0-7 -> K = (L/16)*16 + 0..15.
static __device__ __forceinline__ v16bf load_a_lds(const __bf16* __restrict__ st,
                                                   int row, int k0, int ldk, int hl) {
    const __bf16* p = st + row * ldk + k0 + hl * 16;
    v8bf lo = *reinterpret_cast<const v8bf*>(p);
    v8bf hi = *reinterpret_cast<const v8bf*>(p + 8);
    v16bf a;
#pragma unroll
    for (int t = 0; t < 8; ++t) { a[t] = lo[t]; a[8 + t] = hi[t]; }
    return a;
}

// ---------------- prep: f32 -> bf16 weight copies in workspace ----------------
__global__ void prep_bf16(const float* __restrict__ Wii, const float* __restrict__ Wif,
                          const float* __restrict__ Wig,
                          const float* __restrict__ Whi, const float* __restrict__ Whf,
                          const float* __restrict__ Whc,
                          const float* __restrict__ Whit, const float* __restrict__ Whft,
                          const float* __restrict__ Whct,
                          const float* __restrict__ Vtinv, __bf16* __restrict__ ws) {
    int t = blockIdx.x * blockDim.x + threadIdx.x;  // 0 .. 65535
    __bf16* WiiB = ws;
    __bf16* WifB = WiiB + Hdim * Idim;
    __bf16* WigB = WifB + Hdim * Idim;
    __bf16* WhiB = WigB + Hdim * Idim;
    __bf16* WhfB = WhiB + Hdim * Hdim;
    __bf16* WhcB = WhfB + Hdim * Hdim;
    __bf16* WhitB = WhcB + Hdim * Hdim;
    __bf16* WhftB = WhitB + Hdim * Hdim;
    __bf16* WhctB = WhftB + Hdim * Hdim;
    __bf16* VtTB  = WhctB + Hdim * Hdim;
    if (t < Hdim * Idim) {
        WiiB[t] = (__bf16)Wii[t];
        WifB[t] = (__bf16)Wif[t];
        WigB[t] = (__bf16)Wig[t];
    }
    if (t < Hdim * Hdim) {
        WhiB[t]  = (__bf16)Whi[t];
        WhfB[t]  = (__bf16)Whf[t];
        WhcB[t]  = (__bf16)Whc[t];
        WhitB[t] = (__bf16)Whit[t];
        WhftB[t] = (__bf16)Whft[t];
        WhctB[t] = (__bf16)Whct[t];
        int n = t / Hdim, k = t % Hdim;
        VtTB[(size_t)n * Hdim + k] = (__bf16)Vtinv[(size_t)k * Hdim + n];  // VtT[i][k]=Vtinv[k][i]
    }
}

// ---------------- main fused cell: one workgroup per batch element ----------------
// Cap at 256 VGPRs/lane (4 waves/SIMD) to avoid extended-VGPR msb churn and keep occupancy.
__global__ __launch_bounds__(256, 4)
void spectral_lstm_kernel(const float* __restrict__ x, const float* __restrict__ hx,
                          const float* __restrict__ cx,
                          const float* __restrict__ Wio, const float* __restrict__ Who,
                          const float* __restrict__ Uinv,
                          const float* __restrict__ bi, const float* __restrict__ bf_,
                          const float* __restrict__ bg, const float* __restrict__ bo,
                          const __bf16* __restrict__ ws, float* __restrict__ out) {
    const __bf16* WiiB = ws;
    const __bf16* WifB = WiiB + Hdim * Idim;
    const __bf16* WigB = WifB + Hdim * Idim;
    const __bf16* WhiB = WigB + Hdim * Idim;
    const __bf16* WhfB = WhiB + Hdim * Hdim;
    const __bf16* WhcB = WhfB + Hdim * Hdim;
    const __bf16* WhitB = WhcB + Hdim * Hdim;
    const __bf16* WhftB = WhitB + Hdim * Hdim;
    const __bf16* WhctB = WhftB + Hdim * Hdim;
    const __bf16* VtTB  = WhctB + Hdim * Hdim;

    __shared__ float  s_x[Idim];
    __shared__ float  s_hx[Hdim];
    __shared__ float  s_proj[Hdim];
    __shared__ __bf16 sAx[3 * 16 * Idim];   // scaled Wx strips for current m-tile (12 KB)
    __shared__ __bf16 sAh[3 * 16 * Hdim];   // scaled Wh strips for current m-tile (24 KB)
    __shared__ __bf16 sC[16 * Hdim];        // C-strip for current m-tile (8 KB)

    const int tid  = threadIdx.x;
    const int b    = blockIdx.x;
    const int lane = tid & 31;
    const int l16  = lane & 15;
    const int hl   = lane >> 4;
    const int wave = tid >> 5;

    if (tid < Idim) s_x[tid] = x[(size_t)b * Idim + tid];
    s_hx[tid]   = hx[(size_t)b * Hdim + tid];
    s_proj[tid] = 0.0f;
    __syncthreads();

    // ---- o gate: per-thread dot product with vectorized row loads ----
    float oacc = bo[tid];
    {
        const float4* wx4 = reinterpret_cast<const float4*>(Wio + (size_t)tid * Idim);
#pragma unroll 8
        for (int i = 0; i < Idim / 4; ++i) {
            float4 w = wx4[i];
            oacc += w.x * s_x[4 * i] + w.y * s_x[4 * i + 1] + w.z * s_x[4 * i + 2] + w.w * s_x[4 * i + 3];
        }
        const float4* wh4 = reinterpret_cast<const float4*>(Who + (size_t)tid * Hdim);
#pragma unroll 8
        for (int j = 0; j < Hdim / 4; ++j) {
            float4 w = wh4[j];
            oacc += w.x * s_hx[4 * j] + w.y * s_hx[4 * j + 1] + w.z * s_hx[4 * j + 2] + w.w * s_hx[4 * j + 3];
        }
    }
    const float o_g = sigf(oacc);

    const __bf16* WxA[3]  = {WiiB, WifB, WigB};
    const __bf16* WhA[3]  = {WhiB, WhfB, WhcB};
    const __bf16* WhBt[3] = {WhitB, WhftB, WhctB};
    const float*  biasP[3] = {bi, bf_, bg};

    // whole workgroup cooperates on one m-tile at a time
    for (int mt = 0; mt < 16; ++mt) {
        // ---- phase A: build scaled A-strips in LDS (each element scaled once) ----
#pragma unroll
        for (int g = 0; g < 3; ++g) {
            {   // x part: 16 rows x 128 -> 256 chunks of 8, exactly one per thread
                const int r  = tid >> 4;           // 16 chunks per row
                const int k8 = (tid & 15) * 8;
                v8bf w = *reinterpret_cast<const v8bf*>(WxA[g] + (size_t)(mt * 16 + r) * Idim + k8);
                v8bf s;
#pragma unroll
                for (int t = 0; t < 8; ++t) s[t] = (__bf16)((float)w[t] * s_x[k8 + t]);
                *reinterpret_cast<v8bf*>(sAx + g * (16 * Idim) + r * Idim + k8) = s;
            }
#pragma unroll
            for (int c = 0; c < 2; ++c) {  // h part: 512 chunks of 8 -> two per thread
                const int ch = tid + c * 256;
                const int r  = ch >> 5;            // 32 chunks per row
                const int k8 = (ch & 31) * 8;
                v8bf w = *reinterpret_cast<const v8bf*>(WhA[g] + (size_t)(mt * 16 + r) * Hdim + k8);
                v8bf s;
#pragma unroll
                for (int t = 0; t < 8; ++t) s[t] = (__bf16)((float)w[t] * s_hx[k8 + t]);
                *reinterpret_cast<v8bf*>(sAh + g * (16 * Hdim) + r * Hdim + k8) = s;
            }
        }
        __syncthreads();

        // ---- phase B: each wave computes 2 C tiles (nt = 2*wave, 2*wave+1) ----
#pragma unroll
        for (int p = 0; p < 2; ++p) {
            const int nt   = wave * 2 + p;
            const int rowB = nt * 16 + l16;
            v8f acc[3];
#pragma unroll
            for (int g = 0; g < 3; ++g)
#pragma unroll
                for (int v = 0; v < 8; ++v)
                    acc[g][v] = biasP[g][(size_t)(mt * 16 + v + 8 * hl) * Hdim + nt * 16 + l16];

#pragma unroll
            for (int ks = 0; ks < 4; ++ks) {       // x sandwich: K = 128
#pragma unroll
                for (int g = 0; g < 3; ++g) {
                    v16bf af = load_a_lds(sAx + g * (16 * Idim), l16, ks * 32, Idim, hl);
                    v16bf bf = load_b_frag(WxA[g], rowB, ks * 32, Idim, hl);
                    acc[g] = wmma_bf16(af, bf, acc[g]);
                }
            }
#pragma unroll
            for (int ks = 0; ks < 8; ++ks) {       // hx sandwich: K = 256
#pragma unroll
                for (int g = 0; g < 3; ++g) {
                    v16bf af = load_a_lds(sAh + g * (16 * Hdim), l16, ks * 32, Hdim, hl);
                    v16bf bf = load_b_frag(WhBt[g], rowB, ks * 32, Hdim, hl);
                    acc[g] = wmma_bf16(af, bf, acc[g]);
                }
            }
            // prefetch next m-tile's cx rows
            if (mt + 1 < 16)
                __builtin_prefetch(cx + ((size_t)b * Hdim + (mt + 1) * 16 + l16) * Hdim + nt * 16, 0, 0);

            // fuse: C = sigmoid(f)*cx + sigmoid(i)*tanh(g); store into shared C-strip
#pragma unroll
            for (int v = 0; v < 8; ++v) {
                const int r = v + 8 * hl;                       // row within strip
                const int c = nt * 16 + l16;                    // global column
                const float cxv = cx[((size_t)b * Hdim + mt * 16 + r) * Hdim + c];
                const float iv = sigf(acc[0][v]);
                const float fv = sigf(acc[1][v]);
                const float gv = tanhf(acc[2][v]);
                sC[r * Hdim + c] = (__bf16)(fv * cxv + iv * gv);
            }
        }
        __syncthreads();

        // ---- phase C: T = C_strip @ Vtinv ; each wave owns 2 n'-tiles ----
#pragma unroll
        for (int p = 0; p < 2; ++p) {
            const int np = wave * 2 + p;
            v8f t;
#pragma unroll
            for (int v = 0; v < 8; ++v) t[v] = 0.0f;
#pragma unroll
            for (int ks = 0; ks < 8; ++ks) {
                v16bf aC = load_a_lds(sC, l16, ks * 32, Hdim, hl);
                v16bf bV = load_b_frag(VtTB, np * 16 + l16, ks * 32, Hdim, hl);
                t = wmma_bf16(aC, bV, t);
            }
            // proj[i] += sum_j Uinv[i,j] * T[j,i]   (j in this m-tile)
            const int i = np * 16 + l16;
            float s = 0.0f;
#pragma unroll
            for (int v = 0; v < 8; ++v) {
                const int j = mt * 16 + v + 8 * hl;
                s += Uinv[(size_t)i * Hdim + j] * t[v];
            }
            atomicAdd(&s_proj[i], s);
        }
        __syncthreads();   // protect strips before next m-tile overwrites
    }

    out[(size_t)b * Hdim + tid] = o_g * sigf(s_proj[tid]);
}

extern "C" void kernel_launch(void* const* d_in, const int* in_sizes, int n_in,
                              void* d_out, int out_size, void* d_ws, size_t ws_size,
                              hipStream_t stream) {
    (void)in_sizes; (void)n_in; (void)out_size; (void)ws_size;
    const float* x    = (const float*)d_in[0];
    const float* hx   = (const float*)d_in[1];
    const float* cx   = (const float*)d_in[2];
    const float* Wii  = (const float*)d_in[3];
    const float* Wif  = (const float*)d_in[4];
    const float* Wig  = (const float*)d_in[5];
    const float* Wio  = (const float*)d_in[6];
    const float* Whi  = (const float*)d_in[7];
    const float* Whit = (const float*)d_in[8];
    const float* Whf  = (const float*)d_in[9];
    const float* Whft = (const float*)d_in[10];
    const float* Whc  = (const float*)d_in[11];
    const float* Whct = (const float*)d_in[12];
    const float* Who  = (const float*)d_in[13];
    const float* Uinv = (const float*)d_in[14];
    const float* Vtinv= (const float*)d_in[15];
    const float* bi   = (const float*)d_in[16];
    const float* bf   = (const float*)d_in[17];
    const float* bg   = (const float*)d_in[18];
    const float* bo   = (const float*)d_in[19];
    __bf16* ws = (__bf16*)d_ws;
    float* out = (float*)d_out;

    prep_bf16<<<256, 256, 0, stream>>>(Wii, Wif, Wig, Whi, Whf, Whc, Whit, Whft, Whct, Vtinv, ws);
    spectral_lstm_kernel<<<512, 256, 0, stream>>>(x, hx, cx, Wio, Who, Uinv, bi, bf, bg, bo, ws, out);
}
